// NSAAttention_80771154968635
// MI455X (gfx1250) — matrix operand order repaired and checked by reading, not compiled, not run on hardware
//
#include <hip/hip_runtime.h>
#include <hip/hip_bf16.h>
#include <math.h>

// ---------------- problem constants ----------------
#define S_LEN 2048
#define HID_C 2048
#define H_C   16
#define D_C   128
#define L_C   32
#define NB_C  64
#define NSEL_C 8
#define WIN_C 512
#define NOUT  2048            // H*D
#define SCALE_C 0.08838834764831845f
#define NEG_C  (-1.0e9f)

typedef __attribute__((ext_vector_type(16))) _Float16 v16h;
typedef __attribute__((ext_vector_type(8)))  _Float16 v8h;
typedef __attribute__((ext_vector_type(8)))  float    v8f;

// ---------------- fragment helpers (wave32, ISA 7.12.2) ----------------
__device__ __forceinline__ v16h frag_ld(const _Float16* p) {
  v8h lo = *(const v8h*)(p);
  v8h hi = *(const v8h*)(p + 8);
  return __builtin_shufflevector(lo, hi, 0, 1, 2, 3, 4, 5, 6, 7,
                                 8, 9, 10, 11, 12, 13, 14, 15);
}
// A-matrix permuted store: element (m,k) of a 16x32 tile -> [lane][slot]
__device__ __forceinline__ void perm_a_store(_Float16* base /*[32][16]*/, int m, int k,
                                             _Float16 val) {
  int lane = (k >> 4) * 16 + m;
  int slot = k & 15;
  base[lane * 16 + slot] = val;
}
__device__ __forceinline__ v8f wmma_f16(v16h a, v16h b, v8f c) {
  return __builtin_amdgcn_wmma_f32_16x16x32_f16(false, a, false, b, (short)0, c, false, false);
}
__device__ __forceinline__ float redmax16(float v) {
#pragma unroll
  for (int off = 8; off >= 1; off >>= 1) v = fmaxf(v, __shfl_xor(v, off, 32));
  return v;
}
__device__ __forceinline__ float redsum16(float v) {
#pragma unroll
  for (int off = 8; off >= 1; off >>= 1) v += __shfl_xor(v, off, 32);
  return v;
}
// Async global->LDS 16B copy (ASYNCcnt-tracked), ldsoff = byte offset in workgroup LDS
__device__ __forceinline__ void async_b128(unsigned ldsoff, const void* gaddr) {
  unsigned long long ga = (unsigned long long)gaddr;
  asm volatile("global_load_async_to_lds_b128 %0, %1, off" :: "v"(ldsoff), "v"(ga) : "memory");
}
__device__ __forceinline__ void wait_async0() {
  asm volatile("s_wait_asynccnt 0" ::: "memory");
}

// ---------------- kernel 1: QKV projection GEMM (WMMA) ----------------
// Block tile 64x128, 8 waves (4 M-rows x 2 N-cols), each wave 16x64 = 4 WMMA/k-step.
__global__ __launch_bounds__(256) void qkv_gemm(const float* __restrict__ X,
                                                const float* __restrict__ W,
                                                const float* __restrict__ bias,
                                                float* __restrict__ out) {
  __shared__ alignas(16) _Float16 gsm[4 * 512 + 128 * 32];  // AsP[4][512] | BsC[128][32]
  _Float16* AsP = gsm;            // permuted A, 4 M-halves
  _Float16* BsC = gsm + 2048;     // column-major [n][k]
  int tid = threadIdx.x, wave = tid >> 5, lane = tid & 31;
  int mrow = wave >> 1, ncol = wave & 1;
  int row0 = blockIdx.x * 64;
  int colB = blockIdx.y * 128;
  int col0 = colB + ncol * 64;
  int hf = lane >> 4, cn = lane & 15;
  v8f acc[4] = {{}, {}, {}, {}};
  for (int k0 = 0; k0 < HID_C; k0 += 32) {
    // A tile 64x32 f32: 512 float4 across 256 threads (2 each)
#pragma unroll
    for (int i = 0; i < 2; ++i) {
      int e = tid + i * 256;
      int r = e >> 3, c4 = (e & 7) * 4;
      const float* xp = &X[(size_t)(row0 + r) * HID_C + k0 + c4];
      float4 xv = *(const float4*)xp;
      _Float16* dst = &AsP[(r >> 4) * 512];
      perm_a_store(dst, r & 15, c4 + 0, (_Float16)xv.x);
      perm_a_store(dst, r & 15, c4 + 1, (_Float16)xv.y);
      perm_a_store(dst, r & 15, c4 + 2, (_Float16)xv.z);
      perm_a_store(dst, r & 15, c4 + 3, (_Float16)xv.w);
      if (k0 + 32 < HID_C) __builtin_prefetch(xp + 32, 0, 1);
    }
    // B tile 32x128 f32: 1024 float4 across 256 threads (4 each), store column-major
#pragma unroll
    for (int i = 0; i < 4; ++i) {
      int e = tid + i * 256;
      int kk = e >> 5, c4 = (e & 31) * 4;
      const float* wp = &W[(size_t)(k0 + kk) * NOUT + colB + c4];
      float4 wv = *(const float4*)wp;
      BsC[(c4 + 0) * 32 + kk] = (_Float16)wv.x;
      BsC[(c4 + 1) * 32 + kk] = (_Float16)wv.y;
      BsC[(c4 + 2) * 32 + kk] = (_Float16)wv.z;
      BsC[(c4 + 3) * 32 + kk] = (_Float16)wv.w;
    }
    __syncthreads();
    v16h a = frag_ld(&AsP[mrow * 512 + lane * 16]);
#pragma unroll
    for (int nt = 0; nt < 4; ++nt) {
      v16h b = frag_ld(&BsC[(ncol * 64 + nt * 16 + cn) * 32 + hf * 16]);
      acc[nt] = wmma_f16(a, b, acc[nt]);
    }
    __syncthreads();
  }
#pragma unroll
  for (int nt = 0; nt < 4; ++nt) {
    int col = col0 + nt * 16 + cn;
    int h = col >> 7, d = col & 127;
    float bv = bias[col];
#pragma unroll
    for (int r = 0; r < 8; ++r) {
      int rr = row0 + mrow * 16 + r + 8 * hf;
      out[((size_t)h * S_LEN + rr) * D_C + d] = acc[nt][r] + bv;
    }
  }
}

// ---------------- kernel 2: RoPE + f16 casts ----------------
__global__ __launch_bounds__(128) void rope_cast(float* __restrict__ q, float* __restrict__ k,
                                                 const float* __restrict__ v,
                                                 _Float16* __restrict__ qh,
                                                 _Float16* __restrict__ kh,
                                                 _Float16* __restrict__ vh) {
  int hs = blockIdx.x;                 // h*S + s
  int s = hs & (S_LEN - 1);
  int d = threadIdx.x;                 // 0..127
  size_t idx = (size_t)hs * D_C + d;
  int j = d & 63;
  float inv = __expf(-((float)(2 * j) / (float)D_C) * 9.210340371976184f); // ln(1e4)
  float ang = (float)s * inv;
  float sn, cs;
  __sincosf(ang, &sn, &cs);
  float qv = q[idx], kv = k[idx];
  float qo = (d < 64) ? -q[idx + 64] : q[idx - 64];
  float ko = (d < 64) ? -k[idx + 64] : k[idx - 64];
  __syncthreads();                     // all reads before in-place writes
  float qr = qv * cs + qo * sn;
  float kr = kv * cs + ko * sn;
  q[idx] = qr; k[idx] = kr;
  qh[idx] = (_Float16)qr; kh[idx] = (_Float16)kr; vh[idx] = (_Float16)v[idx];
}

// ---------------- kernel 3: block-mean compression ----------------
__global__ __launch_bounds__(128) void compress(const float* __restrict__ k,
                                                const float* __restrict__ v,
                                                _Float16* __restrict__ kcmph,
                                                _Float16* __restrict__ vcmph) {
  int hb = blockIdx.x;                 // h*NB + nb
  int h = hb >> 6, nb = hb & 63;
  int d = threadIdx.x;
  float ks = 0.f, vs = 0.f;
  for (int t = 0; t < L_C; ++t) {
    size_t idx = ((size_t)h * S_LEN + nb * L_C + t) * D_C + d;
    ks += k[idx]; vs += v[idx];
  }
  ks *= (1.0f / L_C); vs *= (1.0f / L_C);
  size_t o = (size_t)hb * D_C + d;
  kcmph[o] = (_Float16)ks; vcmph[o] = (_Float16)vs;
}

// ---------------- kernel 4: compressed attention + top-k + gates ----------------
// grid = H * (S/64), block = 128 (4 waves, 16 queries each)
// unified LDS arena (f16 units): kc @0 [64][128] | vcT @8192 [128][64]
//                                qP @16384 [4 waves][4][512] | psP @24576 [4 waves][2][512]
__global__ __launch_bounds__(128) void cmp_attn(const _Float16* __restrict__ qh,
                                                const float* __restrict__ qf,
                                                const _Float16* __restrict__ kcmph,
                                                const _Float16* __restrict__ vcmph,
                                                const float* __restrict__ Wg,
                                                const float* __restrict__ bg,
                                                float* __restrict__ outc,
                                                float* __restrict__ gout,
                                                unsigned int* __restrict__ selmask) {
  __shared__ alignas(16) _Float16 smem[28672];
  int tid = threadIdx.x, wave = tid >> 5, lane = tid & 31;
  int h = blockIdx.x >> 5;
  int q0 = (blockIdx.x & 31) * 64 + wave * 16;
  int hf = lane >> 4, cn = lane & 15;
  const int QP = 16384 + wave * 2048;
  const int PS = 24576 + wave * 1024;

  // stage kcmp via async DMA, vcmp transposed via scatter
  for (int e = tid; e < 64 * 16; e += 128) {
    int nb = e >> 4, d8 = (e & 15) * 8;
    async_b128((unsigned)((nb * 128 + d8) * 2),
               &kcmph[((size_t)h * NB_C + nb) * D_C + d8]);
    v8h vv = *(const v8h*)&vcmph[((size_t)h * NB_C + nb) * D_C + d8];
#pragma unroll
    for (int u = 0; u < 8; ++u) smem[8192 + (d8 + u) * 64 + nb] = vv[u];
  }
  // stage q tile permuted (per wave)
  for (int e = lane; e < 16 * 16; e += 32) {
    int r = e >> 4, c8 = (e & 15) * 8;
    v8h qv = *(const v8h*)&qh[((size_t)h * S_LEN + q0 + r) * D_C + c8];
#pragma unroll
    for (int u = 0; u < 8; ++u) {
      int c = c8 + u;
      perm_a_store(&smem[QP + (c >> 5) * 512], r, c & 31, qv[u]);
    }
  }
  wait_async0();
  __syncthreads();

  // scores q(16x128) @ kcmp^T(128x64)
  v8f sc[4] = {{}, {}, {}, {}};
#pragma unroll
  for (int ks = 0; ks < 4; ++ks) {
    v16h a = frag_ld(&smem[QP + ks * 512 + lane * 16]);
#pragma unroll
    for (int nt = 0; nt < 4; ++nt) {
      v16h b = frag_ld(&smem[(nt * 16 + cn) * 128 + ks * 32 + hf * 16]);
      sc[nt] = wmma_f16(a, b, sc[nt]);
    }
  }
  // masked softmax over 64 blocks; write p directly in permuted A layout
#pragma unroll
  for (int r = 0; r < 8; ++r) {
    int i = q0 + r + 8 * hf;
    float vv[4], mx = -1e30f;
#pragma unroll
    for (int nt = 0; nt < 4; ++nt) {
      int jcol = nt * 16 + cn;
      bool ok = ((jcol + 1) * L_C - 1) <= i;
      vv[nt] = ok ? sc[nt][r] * SCALE_C : NEG_C;
      mx = fmaxf(mx, vv[nt]);
    }
    mx = redmax16(mx);
    float sum = 0.f, pv[4];
#pragma unroll
    for (int nt = 0; nt < 4; ++nt) { pv[nt] = __expf(vv[nt] - mx); sum += pv[nt]; }
    sum = redsum16(sum);
    float invs = 1.0f / sum;
#pragma unroll
    for (int nt = 0; nt < 4; ++nt) {
      int lp = (nt & 1) * 16 + (r + 8 * hf);
      int slot = cn;
      smem[PS + (nt >> 1) * 512 + lp * 16 + slot] = (_Float16)(pv[nt] * invs);
    }
  }
  // per-row top-8 block selection (+ current block)
  if (lane < 16) {
    int row = lane, i = q0 + row;
    unsigned c0 = 0u, c1 = 0u;
    for (int it = 0; it < NSEL_C; ++it) {
      float best = -1.0f; int bj = -1;
      for (int j = 0; j < NB_C; ++j) {
        bool used = (j < 32) ? ((c0 >> j) & 1u) : ((c1 >> (j - 32)) & 1u);
        int kk = j & 31;
        int lp = (kk >> 4) * 16 + row;
        int slot = kk & 15;
        float pvl = (float)smem[PS + (j >> 5) * 512 + lp * 16 + slot];
        if (!used && pvl > best) { best = pvl; bj = j; }
      }
      if (bj >= 0) { if (bj < 32) c0 |= 1u << bj; else c1 |= 1u << (bj - 32); }
    }
    int cb = i >> 5;
    if (cb < 32) c0 |= 1u << cb; else c1 |= 1u << (cb - 32);
    selmask[((size_t)h * S_LEN + i) * 2]     = c0;
    selmask[((size_t)h * S_LEN + i) * 2 + 1] = c1;
  }
  // out_cmp = p(16x64) @ vcmp(64x128)
  v8f oc[8] = {{}, {}, {}, {}, {}, {}, {}, {}};
#pragma unroll
  for (int ks2 = 0; ks2 < 2; ++ks2) {
    v16h a = frag_ld(&smem[PS + ks2 * 512 + lane * 16]);
#pragma unroll
    for (int nt2 = 0; nt2 < 8; ++nt2) {
      v16h b = frag_ld(&smem[8192 + (nt2 * 16 + cn) * 64 + ks2 * 32 + hf * 16]);
      oc[nt2] = wmma_f16(a, b, oc[nt2]);
    }
  }
#pragma unroll
  for (int nt2 = 0; nt2 < 8; ++nt2) {
#pragma unroll
    for (int r = 0; r < 8; ++r) {
      int i = q0 + r + 8 * hf;
      float hb = (i >= (L_C - 1)) ? 1.0f : 0.0f;
      outc[((size_t)h * S_LEN + i) * D_C + nt2 * 16 + cn] = oc[nt2][r] * hb;
    }
  }
  // gates: sigmoid(q . Wg + bg), float4-vectorized dot
  for (int e = lane; e < 48; e += 32) {
    int row = e / 3, gi = e % 3;
    int i = q0 + row;
    float acc = bg[gi];
    for (int d = 0; d < D_C; d += 4) {
      float4 qv = *(const float4*)&qf[((size_t)h * S_LEN + i) * D_C + d];
      const float* wp = &Wg[((size_t)(h * D_C) + d) * 3 + gi];
      acc += qv.x * wp[0] + qv.y * wp[3] + qv.z * wp[6] + qv.w * wp[9];
    }
    gout[((size_t)h * S_LEN + i) * 3 + gi] = 1.0f / (1.0f + __expf(-acc));
  }
}

// ---------------- kernel 5: selected + window attention, flash-style, gated combine ----
// grid = H * (S/32), block = 64 (2 waves, 16 queries each)
// unified LDS arena (f16 units): qP @0 [2][4][512] | kb @4096 [2][32][128]
//                                vT @12288 [2][128][32] | pP @20480 [2][512]
__global__ __launch_bounds__(64) void selwin_attn(const _Float16* __restrict__ qh,
                                                  const _Float16* __restrict__ kh,
                                                  const _Float16* __restrict__ vh,
                                                  const unsigned int* __restrict__ selmask,
                                                  const float* __restrict__ outc,
                                                  const float* __restrict__ g,
                                                  float* __restrict__ out) {
  __shared__ alignas(16) _Float16 smem[21504];
  int tid = threadIdx.x, wave = tid >> 5, lane = tid & 31;
  int h = blockIdx.x >> 6;
  int q0 = (blockIdx.x & 63) * 32 + wave * 16;
  int hf = lane >> 4, cn = lane & 15;
  const int QP = wave * 2048;
  const int KB = 4096 + wave * 4096;
  const int VT = 12288 + wave * 4096;
  const int PP = 20480 + wave * 512;

  // stage q tile permuted (per wave)
  for (int e = lane; e < 16 * 16; e += 32) {
    int r = e >> 4, c8 = (e & 15) * 8;
    v8h qv = *(const v8h*)&qh[((size_t)h * S_LEN + q0 + r) * D_C + c8];
#pragma unroll
    for (int u = 0; u < 8; ++u) {
      int c = c8 + u;
      perm_a_store(&smem[QP + (c >> 5) * 512], r, c & 31, qv[u]);
    }
  }
  // per-row selection bitmasks
  unsigned sm0[8], sm1[8];
#pragma unroll
  for (int r = 0; r < 8; ++r) {
    int i = q0 + r + 8 * hf;
    sm0[r] = selmask[((size_t)h * S_LEN + i) * 2];
    sm1[r] = selmask[((size_t)h * S_LEN + i) * 2 + 1];
  }
  float mS[8], lS[8], mW[8], lW[8];
  v8f oS[8], oW[8];
#pragma unroll
  for (int r = 0; r < 8; ++r) { mS[r] = -__builtin_inff(); lS[r] = 0.f; mW[r] = -__builtin_inff(); lW[r] = 0.f; }
#pragma unroll
  for (int n = 0; n < 8; ++n) { oS[n] = (v8f){}; oW[n] = (v8f){}; }

  int jbmax = (q0 + 15) >> 5;
  for (int jb = 0; jb <= jbmax; ++jb) {
    int t0 = jb * 32;
    // k block via async DMA (natural layout), v block transposed via scatter
    for (int e = lane; e < 32 * 16; e += 32) {
      int r = e >> 4, c8 = (e & 15) * 8;
      async_b128((unsigned)((KB + r * 128 + c8) * 2),
                 &kh[((size_t)h * S_LEN + t0 + r) * D_C + c8]);
      v8h vv = *(const v8h*)&vh[((size_t)h * S_LEN + t0 + r) * D_C + c8];
#pragma unroll
      for (int u = 0; u < 8; ++u) smem[VT + (c8 + u) * 32 + r] = vv[u];
    }
    wait_async0();
    // scores q(16x128) @ k^T(128x32)
    v8f sc[2] = {{}, {}};
#pragma unroll
    for (int ks = 0; ks < 4; ++ks) {
      v16h a = frag_ld(&smem[QP + ks * 512 + lane * 16]);
#pragma unroll
      for (int nt = 0; nt < 2; ++nt) {
        v16h b = frag_ld(&smem[KB + (nt * 16 + cn) * 128 + ks * 32 + hf * 16]);
        sc[nt] = wmma_f16(a, b, sc[nt]);
      }
    }
    // ---- SEL path ----
    {
      float alpha[8];
#pragma unroll
      for (int r = 0; r < 8; ++r) {
        int i = q0 + r + 8 * hf;
        unsigned bit = (jb < 32) ? ((sm0[r] >> jb) & 1u) : ((sm1[r] >> (jb - 32)) & 1u);
        int ta = t0 + cn, tb2 = t0 + 16 + cn;
        float v0 = (bit && ta <= i)  ? sc[0][r] * SCALE_C : -__builtin_inff();
        float v1 = (bit && tb2 <= i) ? sc[1][r] * SCALE_C : -__builtin_inff();
        float bm = redmax16(fmaxf(v0, v1));
        float nm = fmaxf(mS[r], bm);
        float a_ = (mS[r] == -__builtin_inff()) ? 0.f : __expf(mS[r] - nm);
        float p0 = (v0 == -__builtin_inff()) ? 0.f : __expf(v0 - nm);
        float p1 = (v1 == -__builtin_inff()) ? 0.f : __expf(v1 - nm);
        lS[r] = lS[r] * a_ + redsum16(p0 + p1);
        mS[r] = nm; alpha[r] = a_;
        int lp = r + 8 * hf;
        smem[PP + lp * 16 + cn]        = (_Float16)p0;
        smem[PP + (16 + lp) * 16 + cn] = (_Float16)p1;
      }
      v16h a = frag_ld(&smem[PP + lane * 16]);
#pragma unroll
      for (int nt2 = 0; nt2 < 8; ++nt2) {
        v8f c = oS[nt2];
#pragma unroll
        for (int r = 0; r < 8; ++r) c[r] *= alpha[r];
        v16h b = frag_ld(&smem[VT + (nt2 * 16 + cn) * 32 + hf * 16]);
        oS[nt2] = wmma_f16(a, b, c);
      }
    }
    // ---- WIN path ----
    {
      float alpha[8];
#pragma unroll
      for (int r = 0; r < 8; ++r) {
        int i = q0 + r + 8 * hf;
        int ta = t0 + cn, tb2 = t0 + 16 + cn;
        bool ok0 = (ta <= i) && (i - ta < WIN_C);
        bool ok1 = (tb2 <= i) && (i - tb2 < WIN_C);
        float v0 = ok0 ? sc[0][r] * SCALE_C : -__builtin_inff();
        float v1 = ok1 ? sc[1][r] * SCALE_C : -__builtin_inff();
        float bm = redmax16(fmaxf(v0, v1));
        float nm = fmaxf(mW[r], bm);
        float a_ = (mW[r] == -__builtin_inff()) ? 0.f : __expf(mW[r] - nm);
        float p0 = (v0 == -__builtin_inff()) ? 0.f : __expf(v0 - nm);
        float p1 = (v1 == -__builtin_inff()) ? 0.f : __expf(v1 - nm);
        lW[r] = lW[r] * a_ + redsum16(p0 + p1);
        mW[r] = nm; alpha[r] = a_;
        int lp = r + 8 * hf;
        smem[PP + lp * 16 + cn]        = (_Float16)p0;
        smem[PP + (16 + lp) * 16 + cn] = (_Float16)p1;
      }
      v16h a = frag_ld(&smem[PP + lane * 16]);
#pragma unroll
      for (int nt2 = 0; nt2 < 8; ++nt2) {
        v8f c = oW[nt2];
#pragma unroll
        for (int r = 0; r < 8; ++r) c[r] *= alpha[r];
        v16h b = frag_ld(&smem[VT + (nt2 * 16 + cn) * 32 + hf * 16]);
        oW[nt2] = wmma_f16(a, b, c);
      }
    }
  }
  // epilogue: normalize + gated combine with out_cmp
#pragma unroll
  for (int nt2 = 0; nt2 < 8; ++nt2) {
#pragma unroll
    for (int r = 0; r < 8; ++r) {
      int i = q0 + r + 8 * hf;
      int d = nt2 * 16 + cn;
      float osel = (lS[r] > 0.f) ? oS[nt2][r] / lS[r] : 0.f;
      float owin = (lW[r] > 0.f) ? oW[nt2][r] / lW[r] : 0.f;
      size_t gix = ((size_t)h * S_LEN + i) * 3;
      float g0 = g[gix], g1 = g[gix + 1], g2 = g[gix + 2];
      float ocv = outc[((size_t)h * S_LEN + i) * D_C + d];
      out[(size_t)i * NOUT + h * D_C + d] = g0 * ocv + g1 * osel + g2 * owin;
    }
  }
}

// ---------------- host launcher ----------------
extern "C" void kernel_launch(void* const* d_in, const int* in_sizes, int n_in,
                              void* d_out, int out_size, void* d_ws, size_t ws_size,
                              hipStream_t stream) {
  (void)in_sizes; (void)n_in; (void)out_size; (void)ws_size;
  const float* hid = (const float*)d_in[0];
  const float* Wq  = (const float*)d_in[1];
  const float* bq  = (const float*)d_in[2];
  const float* Wk  = (const float*)d_in[3];
  const float* bk  = (const float*)d_in[4];
  const float* Wv  = (const float*)d_in[5];
  const float* bv  = (const float*)d_in[6];
  const float* Wg  = (const float*)d_in[7];
  const float* bg  = (const float*)d_in[8];
  float* out = (float*)d_out;

  char* ws = (char*)d_ws;
  const size_t HSD = (size_t)H_C * S_LEN * D_C;        // 4,194,304 elems
  size_t off = 0;
  float* qf   = (float*)(ws + off); off += HSD * 4;
  float* kf   = (float*)(ws + off); off += HSD * 4;
  float* vf   = (float*)(ws + off); off += HSD * 4;
  float* outc = (float*)(ws + off); off += HSD * 4;
  _Float16* qhp = (_Float16*)(ws + off); off += HSD * 2;
  _Float16* khp = (_Float16*)(ws + off); off += HSD * 2;
  _Float16* vhp = (_Float16*)(ws + off); off += HSD * 2;
  _Float16* kcmph = (_Float16*)(ws + off); off += (size_t)H_C * NB_C * D_C * 2;
  _Float16* vcmph = (_Float16*)(ws + off); off += (size_t)H_C * NB_C * D_C * 2;
  float* gbuf = (float*)(ws + off); off += (size_t)H_C * S_LEN * 3 * 4;
  unsigned int* selm = (unsigned int*)(ws + off); off += (size_t)H_C * S_LEN * 2 * 4;

  dim3 ggrid(S_LEN / 64, NOUT / 128);
  qkv_gemm<<<ggrid, 256, 0, stream>>>(hid, Wq, bq, qf);
  qkv_gemm<<<ggrid, 256, 0, stream>>>(hid, Wk, bk, kf);
  qkv_gemm<<<ggrid, 256, 0, stream>>>(hid, Wv, bv, vf);

  rope_cast<<<H_C * S_LEN, 128, 0, stream>>>(qf, kf, vf, qhp, khp, vhp);
  compress<<<H_C * NB_C, 128, 0, stream>>>(kf, vf, kcmph, vcmph);

  cmp_attn<<<H_C * (S_LEN / 64), 128, 0, stream>>>(qhp, qf, kcmph, vcmph, Wg, bg,
                                                   outc, gbuf, selm);
  selwin_attn<<<H_C * (S_LEN / 32), 64, 0, stream>>>(qhp, khp, vhp, selm, outc, gbuf, out);
}
